// ICON_Transformer_21758304322137
// MI455X (gfx1250) — compile-verified, matrix operand order checked by reading, not run on hardware
//
#include <hip/hip_runtime.h>
#include <hip/hip_bf16.h>

// MI455X (gfx1250) implementation.
// One workgroup per cell (s = B*N = 16384). 512 threads = 16 wave32s.
// All GEMMs use v_wmma_f32_16x16x32_f16 (M=16 rows == neighborhood size).
// Memory-bound problem (~560MB traffic vs ~213 GFLOP) -> f16 compute is free,
// stage everything in LDS, weights converted once to f16 in d_ws (L2-resident).

typedef __attribute__((ext_vector_type(16))) _Float16 v16h;
typedef __attribute__((ext_vector_type(8)))  float    v8f;

#define WMMA_F16(a, b, c) \
    __builtin_amdgcn_wmma_f32_16x16x32_f16(false, (a), false, (b), (short)0, (c), false, false)

// ---------------------------------------------------------------------------
// Pre-kernel: convert one 256x256 f32 weight matrix to f16, transposed so
// WT[n*256 + k] = W[k*256 + n]  (WMMA B-fragment reads contiguous K per lane).
// ---------------------------------------------------------------------------
__global__ void wconv_kernel(const float* __restrict__ W, _Float16* __restrict__ WT) {
    int e = blockIdx.x * 256 + threadIdx.x;   // 65536 elements
    int i = e >> 8;        // row of W (k)
    int j = e & 255;       // col of W (n)
    WT[j * 256 + i] = (_Float16)W[e];
}

// ---------------------------------------------------------------------------
// Main kernel: one block per cell.
// ---------------------------------------------------------------------------
__launch_bounds__(512)
__global__ void icon_cell_kernel(
    const float* __restrict__ x,      // [s,16,256]
    const float* __restrict__ dist,   // [s,16,16]
    const float* __restrict__ phi,    // [s,16,16]
    const unsigned char* __restrict__ mask, // [s,16,16] bool
    const float* __restrict__ bq, const float* __restrict__ bk,
    const float* __restrict__ bv, const float* __restrict__ bo,
    const float* __restrict__ Wpos,   // [2,16]
    const float* __restrict__ Wbias,  // [16,8]
    const float* __restrict__ g1, const float* __restrict__ b1,
    const float* __restrict__ g2, const float* __restrict__ b2,
    const _Float16* __restrict__ wT,  // 6 x [256][256] f16 transposed: q,k,v,o,1,2
    float* __restrict__ out)          // [s,16,256]
{
    constexpr int SX = 272;               // f16 row stride (17*16 -> 32B aligned, bank-spread)
    const int cell = blockIdx.x;
    const int tid  = threadIdx.x;
    const int wave = tid >> 5;
    const int lane = tid & 31;
    const int laneN = lane & 15;          // N (or M for A-frag) index within fragment
    const int hs    = lane >> 4;          // which 16-wide K-half this lane holds

    __shared__ alignas(32) _Float16 lds_x[16 * SX];  // x(f16), later h1(f16)
    __shared__ alignas(32) _Float16 lds_q[16 * SX];  // q, later silu(h@W1)
    __shared__ alignas(32) _Float16 lds_k[16 * SX];
    __shared__ alignas(32) _Float16 lds_o[16 * SX];  // attention output
    __shared__ alignas(32) unsigned char pool[24576]; // {vT[256][32] + att[8][16][32]} U hbuf f32[16][256]
    __shared__ float sWpos[32];
    __shared__ float sWbias[128];

    _Float16* vt   = (_Float16*)pool;              // vT[d][key], keys 16..31 zero pad
    _Float16* attb = (_Float16*)(pool + 16384);    // att[h][q][k], k 16..31 zero pad
    float*    hbuf = (float*)pool;                 // f32 [16][256] residual / LN buffer

    // ---- Phase 0: zero pads, load x, load tiny tables -----------------------
    {
        unsigned int* pz = (unsigned int*)pool;
        for (int i = tid; i < 6144; i += 512) pz[i] = 0u;
        const float* xc = x + (size_t)cell * 4096;
        for (int i = tid; i < 4096; i += 512)
            lds_x[(i >> 8) * SX + (i & 255)] = (_Float16)xc[i];
        if (tid < 32)  sWpos[tid]  = Wpos[tid];
        if (tid < 128) sWbias[tid] = Wbias[tid];
    }
    __syncthreads();

    // ---- Phase 1: Q/K/V projections (16 waves, one 16-col tile each) --------
    {
        const int nt = wave;
        for (int m = 0; m < 3; ++m) {
            const _Float16* WT = wT + m * 65536;
            const float* bias = (m == 0) ? bq : (m == 1) ? bk : bv;
            float bb = bias[nt * 16 + laneN];
            v8f c;
            #pragma unroll
            for (int r = 0; r < 8; ++r) c[r] = bb;
            #pragma unroll
            for (int ks = 0; ks < 8; ++ks) {
                v16h a = *(const v16h*)&lds_x[laneN * SX + ks * 32 + hs * 16];
                v16h b = *(const v16h*)&WT[(nt * 16 + laneN) * 256 + ks * 32 + hs * 16];
                c = WMMA_F16(a, b, c);
            }
            #pragma unroll
            for (int r = 0; r < 8; ++r) {
                int M = r + 8 * hs;
                int col = nt * 16 + laneN;
                if (m == 0)      lds_q[M * SX + col] = (_Float16)c[r];
                else if (m == 1) lds_k[M * SX + col] = (_Float16)c[r];
                else             vt[col * 32 + M]    = (_Float16)c[r]; // transposed V
            }
        }
    }
    __syncthreads();

    // ---- Phase 2: per-head attention (waves 0..7, head = wave) --------------
    if (wave < 8) {
        const int h = wave;
        v8f z = {};
        // scores = q . k^T  (A: q[16x32], B: k^T[32x16] = contiguous rows of k)
        v16h aq = *(const v16h*)&lds_q[laneN * SX + h * 32 + hs * 16];
        v16h bkf = *(const v16h*)&lds_k[laneN * SX + h * 32 + hs * 16];
        v8f s = WMMA_F16(aq, bkf, z);

        const float scale = 0.17677669529663687f;   // 1/sqrt(32)
        const float* dC = dist + (size_t)cell * 256;
        const float* pC = phi  + (size_t)cell * 256;
        const unsigned char* mC = mask + (size_t)cell * 256;
        #pragma unroll
        for (int r = 0; r < 8; ++r) {
            int qi = r + 8 * hs;
            int off = qi * 16 + laneN;
            float sv = s[r] * scale;
            float dd = dC[off], pp = pC[off];
            float bsum = 0.f;
            #pragma unroll
            for (int j = 0; j < 16; ++j) {
                float t = dd * sWpos[j] + pp * sWpos[16 + j];
                t = t * (1.0f / (1.0f + __expf(-t)));   // silu
                bsum += t * sWbias[j * 8 + h];
            }
            sv += bsum;
            if (mC[off]) sv = -1e9f;
            s[r] = sv;
        }
        // softmax over keys: 16 lanes of each half-group hold one row
        #pragma unroll
        for (int r = 0; r < 8; ++r) {
            float m0 = s[r];
            m0 = fmaxf(m0, __shfl_xor(m0, 1));
            m0 = fmaxf(m0, __shfl_xor(m0, 2));
            m0 = fmaxf(m0, __shfl_xor(m0, 4));
            m0 = fmaxf(m0, __shfl_xor(m0, 8));
            float e = __expf(s[r] - m0);
            float su = e;
            su += __shfl_xor(su, 1);
            su += __shfl_xor(su, 2);
            su += __shfl_xor(su, 4);
            su += __shfl_xor(su, 8);
            attb[(h * 16 + (r + 8 * hs)) * 32 + laneN] = (_Float16)(e / su);
        }
        // out = att @ v   (K = 16 keys zero-padded to 32; B from transposed V)
        #pragma unroll
        for (int nt2 = 0; nt2 < 2; ++nt2) {
            v16h aa = *(const v16h*)&attb[(h * 16 + laneN) * 32 + hs * 16];
            v16h bb = *(const v16h*)&vt[(h * 32 + nt2 * 16 + laneN) * 32 + hs * 16];
            v8f o = WMMA_F16(aa, bb, z);
            #pragma unroll
            for (int r = 0; r < 8; ++r)
                lds_o[(r + 8 * hs) * SX + h * 32 + nt2 * 16 + laneN] = (_Float16)o[r];
        }
    }
    __syncthreads();

    // ---- Phase 3: Wo projection + residual (hbuf overwrites vt/att region) --
    {
        const int nt = wave;
        const _Float16* WT = wT + 3 * 65536;
        float bb = bo[nt * 16 + laneN];
        v8f c;
        #pragma unroll
        for (int r = 0; r < 8; ++r) c[r] = bb;
        #pragma unroll
        for (int ks = 0; ks < 8; ++ks) {
            v16h a = *(const v16h*)&lds_o[laneN * SX + ks * 32 + hs * 16];
            v16h b = *(const v16h*)&WT[(nt * 16 + laneN) * 256 + ks * 32 + hs * 16];
            c = WMMA_F16(a, b, c);
        }
        #pragma unroll
        for (int r = 0; r < 8; ++r) {
            int M = r + 8 * hs;
            int col = nt * 16 + laneN;
            hbuf[M * 256 + col] = (float)lds_x[M * SX + col] + c[r];  // residual
        }
    }
    __syncthreads();

    // ---- Phase 4: LayerNorm 1 (one wave per row) ----------------------------
    {
        const int row = wave;
        float vals[8], sum = 0.f, sq = 0.f;
        #pragma unroll
        for (int i = 0; i < 8; ++i) {
            float v = hbuf[row * 256 + lane + i * 32];
            vals[i] = v; sum += v; sq += v * v;
        }
        #pragma unroll
        for (int d = 16; d >= 1; d >>= 1) { sum += __shfl_xor(sum, d); sq += __shfl_xor(sq, d); }
        float mean = sum * (1.0f / 256.0f);
        float var  = sq  * (1.0f / 256.0f) - mean * mean;
        float rstd = rsqrtf(var + 1e-5f);
        #pragma unroll
        for (int i = 0; i < 8; ++i) {
            int c0 = lane + i * 32;
            float nv = (vals[i] - mean) * rstd * g1[c0] + b1[c0];
            hbuf[row * 256 + c0] = nv;              // h1 (f32 residual source)
            lds_x[row * SX + c0] = (_Float16)nv;    // h1 (f16 GEMM input)
        }
    }
    __syncthreads();

    // ---- Phase 5: MLP first GEMM + SiLU -> lds_q ----------------------------
    {
        const int nt = wave;
        const _Float16* WT = wT + 4 * 65536;
        v8f c = {};
        #pragma unroll
        for (int ks = 0; ks < 8; ++ks) {
            v16h a = *(const v16h*)&lds_x[laneN * SX + ks * 32 + hs * 16];
            v16h b = *(const v16h*)&WT[(nt * 16 + laneN) * 256 + ks * 32 + hs * 16];
            c = WMMA_F16(a, b, c);
        }
        #pragma unroll
        for (int r = 0; r < 8; ++r) {
            float t = c[r];
            t = t * (1.0f / (1.0f + __expf(-t)));   // silu
            lds_q[(r + 8 * hs) * SX + nt * 16 + laneN] = (_Float16)t;
        }
    }
    __syncthreads();

    // ---- Phase 6: MLP second GEMM + residual into hbuf ----------------------
    {
        const int nt = wave;
        const _Float16* WT = wT + 5 * 65536;
        v8f c = {};
        #pragma unroll
        for (int ks = 0; ks < 8; ++ks) {
            v16h a = *(const v16h*)&lds_q[laneN * SX + ks * 32 + hs * 16];
            v16h b = *(const v16h*)&WT[(nt * 16 + laneN) * 256 + ks * 32 + hs * 16];
            c = WMMA_F16(a, b, c);
        }
        #pragma unroll
        for (int r = 0; r < 8; ++r) {
            int M = r + 8 * hs;
            hbuf[M * 256 + nt * 16 + laneN] += c[r];
        }
    }
    __syncthreads();

    // ---- Phase 7: LayerNorm 2 -> global output ------------------------------
    {
        const int row = wave;
        float vals[8], sum = 0.f, sq = 0.f;
        #pragma unroll
        for (int i = 0; i < 8; ++i) {
            float v = hbuf[row * 256 + lane + i * 32];
            vals[i] = v; sum += v; sq += v * v;
        }
        #pragma unroll
        for (int d = 16; d >= 1; d >>= 1) { sum += __shfl_xor(sum, d); sq += __shfl_xor(sq, d); }
        float mean = sum * (1.0f / 256.0f);
        float var  = sq  * (1.0f / 256.0f) - mean * mean;
        float rstd = rsqrtf(var + 1e-5f);
        float* oc = out + (size_t)cell * 4096 + row * 256;
        #pragma unroll
        for (int i = 0; i < 8; ++i) {
            int c0 = lane + i * 32;
            oc[c0] = (vals[i] - mean) * rstd * g2[c0] + b2[c0];
        }
    }
}

// ---------------------------------------------------------------------------
extern "C" void kernel_launch(void* const* d_in, const int* in_sizes, int n_in,
                              void* d_out, int out_size, void* d_ws, size_t ws_size,
                              hipStream_t stream) {
    const float* x    = (const float*)d_in[0];
    const float* dist = (const float*)d_in[1];
    const float* phi  = (const float*)d_in[2];
    const unsigned char* mask = (const unsigned char*)d_in[3];
    const float* bq = (const float*)d_in[5];
    const float* bk = (const float*)d_in[7];
    const float* bv = (const float*)d_in[9];
    const float* bo = (const float*)d_in[11];
    const float* Wpos  = (const float*)d_in[12];
    const float* Wbias = (const float*)d_in[13];
    const float* g1 = (const float*)d_in[16];
    const float* b1 = (const float*)d_in[17];
    const float* g2 = (const float*)d_in[18];
    const float* b2 = (const float*)d_in[19];

    _Float16* wT = (_Float16*)d_ws;   // 6 x 65536 f16 = 768 KB
    const float* Ws[6] = {
        (const float*)d_in[4],   // Wq
        (const float*)d_in[6],   // Wk
        (const float*)d_in[8],   // Wv
        (const float*)d_in[10],  // Wo
        (const float*)d_in[14],  // W1
        (const float*)d_in[15],  // W2
    };
    for (int m = 0; m < 6; ++m)
        wconv_kernel<<<256, 256, 0, stream>>>(Ws[m], wT + m * 65536);

    int s = in_sizes[0] >> 12;   // B*N = numel(x) / (16*256)
    icon_cell_kernel<<<s, 512, 0, stream>>>(
        x, dist, phi, mask, bq, bk, bv, bo, Wpos, Wbias,
        g1, b1, g2, b2, wT, (float*)d_out);
}
